// FanoOctonionAttention_43224550867048
// MI455X (gfx1250) — matrix operand order repaired and checked by reading, compiled, not run on hardware
//
#include <hip/hip_runtime.h>

typedef __attribute__((ext_vector_type(16))) _Float16 v16h;
typedef __attribute__((ext_vector_type(8)))  _Float16 v8h;
typedef __attribute__((ext_vector_type(8)))  float    v8f;

static constexpr int T_    = 2048;
static constexpr int BATCH = 4;
static constexpr int HID   = 896;
static constexpr int HD    = 128;
static constexpr int NH    = 7;
static constexpr int MTOK  = BATCH * T_;          // 8192 token-rows

// ---- workspace layout (units: _Float16 elements) ----
static constexpr size_t SZ_XH  = (size_t)MTOK * HID;     // 7,340,032
static constexpr size_t SZ_W   = (size_t)NH * HID * HD;  //   802,816
static constexpr size_t SZ_WO  = (size_t)HID * HID;      //   802,816
static constexpr size_t SZ_QKV = (size_t)NH * MTOK * HD; // 7,340,032
static constexpr size_t OFF_XH  = 0;
static constexpr size_t OFF_WQ  = OFF_XH  + SZ_XH;
static constexpr size_t OFF_WK  = OFF_WQ  + SZ_W;
static constexpr size_t OFF_WV  = OFF_WK  + SZ_W;
static constexpr size_t OFF_WOT = OFF_WV  + SZ_W;
static constexpr size_t OFF_Q   = OFF_WOT + SZ_WO;
static constexpr size_t OFF_K   = OFF_Q   + SZ_QKV;
static constexpr size_t OFF_VT  = OFF_K   + SZ_QKV;
static constexpr size_t OFF_CC  = OFF_VT  + SZ_QKV;

static __device__ __forceinline__ v16h cat16(v8h lo, v8h hi) {
  return __builtin_shufflevector(lo, hi, 0,1,2,3,4,5,6,7,8,9,10,11,12,13,14,15);
}
static __device__ __forceinline__ v8f wmma16(v16h a, v16h b, v8f c) {
  return __builtin_amdgcn_wmma_f32_16x16x32_f16(false, a, false, b, (short)0, c, false, false);
}

// Async DMA: global -> LDS, 16 bytes per lane, tracked by ASYNCcnt.
// Flat pointers to __shared__ carry the LDS byte offset in their low 32 bits.
static __device__ __forceinline__ void async_b128(void* lds_dst, const _Float16* gsrc) {
  unsigned loff = (unsigned)(uintptr_t)lds_dst;
  asm volatile("global_load_async_to_lds_b128 %0, %1, off"
               :: "v"(loff), "v"(gsrc) : "memory");
}
static __device__ __forceinline__ void wait_async0() {
  asm volatile("s_wait_asynccnt 0x0" ::: "memory");
}

// ---------------- conversion kernels ----------------
__global__ void cvt_f32_to_f16(const float* __restrict__ s, _Float16* __restrict__ d, int n) {
  int i = blockIdx.x * blockDim.x + threadIdx.x;
  if (i < n) d[i] = (_Float16)s[i];
}
__global__ void wo_transpose(const float* __restrict__ Wo, _Float16* __restrict__ WoT) {
  int i = blockIdx.x * blockDim.x + threadIdx.x;
  if (i < HID * HID) {
    int k = i / HID, n = i % HID;               // WoT[k][n] = Wo[n][k]
    WoT[i] = (_Float16)Wo[(size_t)n * HID + k];
  }
}

// ---------------- WMMA GEMM: C = A(MxK) * B(KxN), f16 in, f32 acc ----------------
// OUTMODE: 0 = f16 row-major, 1 = f16 transposed (C^T, ld = ldo), 2 = f32 row-major
template <int OUTMODE>
__global__ __launch_bounds__(256)
void gemm16(const _Float16* __restrict__ A, const _Float16* __restrict__ Bbase,
            size_t strideB, void* __restrict__ Obase, size_t strideO,
            int Kd, int N, int ldo, int nblocks) {
  __shared__ __align__(16) _Float16 As[128 * 40];
  __shared__ __align__(16) _Float16 Bt[64 * 40];

  const int z  = blockIdx.z;
  const _Float16* B = Bbase + strideB * (size_t)z;
  const int mb = blockIdx.x / nblocks;
  const int nb = blockIdx.x % nblocks;
  const int mbase = mb * 128, nbase = nb * 64;
  const int tid = threadIdx.x, lane = tid & 31, w = tid >> 5;
  const int g = lane >> 4, ln = lane & 15;

  v8f acc[4];
#pragma unroll
  for (int i = 0; i < 4; ++i)
#pragma unroll
    for (int r = 0; r < 8; ++r) acc[i][r] = 0.f;

  for (int kc = 0; kc < Kd; kc += 32) {
    { // async-stage A tile: 128 rows x 32 halves (8 KB, 2 x b128 per thread)
#pragma unroll
      for (int i = 0; i < 2; ++i) {
        int c = tid * 2 + i;                    // 0..511
        int row = c >> 2, col = (c & 3) * 8;
        async_b128(&As[row * 40 + col], A + (size_t)(mbase + row) * Kd + kc + col);
      }
    }
    { // stage B^T tile: Bt[n][k], 64 x 32
#pragma unroll
      for (int i = 0; i < 4; ++i) {
        int idx = tid * 4 + i;                  // 0..1023
        int k   = idx >> 5;                     // 0..31
        int np  = (idx & 31) * 2;               // 0..62
        unsigned d = *(const unsigned*)(B + (size_t)(kc + k) * N + nbase + np);
        Bt[(np)     * 40 + k] = ((const _Float16*)&d)[0];
        Bt[(np + 1) * 40 + k] = ((const _Float16*)&d)[1];
      }
    }
    wait_async0();
    __syncthreads();

    v16h aop;
    {
      int row = w * 16 + ln, off = g * 8;
      v8h lo = *(v8h*)(&As[row * 40 + off]);
      v8h hi = *(v8h*)(&As[row * 40 + 16 + off]);
      aop = cat16(lo, hi);
    }
#pragma unroll
    for (int nt = 0; nt < 4; ++nt) {
      int n = nt * 16 + ln, koff = g * 8;
      v8h lo = *(v8h*)(&Bt[n * 40 + koff]);
      v8h hi = *(v8h*)(&Bt[n * 40 + koff + 16]);
      acc[nt] = wmma16(aop, cat16(lo, hi), acc[nt]);
    }
    __syncthreads();
  }

  if constexpr (OUTMODE == 0) {
    _Float16* O = (_Float16*)Obase + strideO * (size_t)z;
#pragma unroll
    for (int nt = 0; nt < 4; ++nt)
#pragma unroll
      for (int r = 0; r < 8; ++r)
        O[(size_t)(mbase + w * 16 + r + 8 * g) * ldo + nbase + nt * 16 + ln] =
            (_Float16)acc[nt][r];
  } else if constexpr (OUTMODE == 1) {
    _Float16* O = (_Float16*)Obase + strideO * (size_t)z;
#pragma unroll
    for (int nt = 0; nt < 4; ++nt) {
      v8h pk;
#pragma unroll
      for (int r = 0; r < 8; ++r) pk[r] = (_Float16)acc[nt][r];
      *(v8h*)(O + (size_t)(nbase + nt * 16 + ln) * ldo + mbase + w * 16 + 8 * g) = pk;
    }
  } else {
    float* O = (float*)Obase + strideO * (size_t)z;
#pragma unroll
    for (int nt = 0; nt < 4; ++nt)
#pragma unroll
      for (int r = 0; r < 8; ++r)
        O[(size_t)(mbase + w * 16 + r + 8 * g) * ldo + nbase + nt * 16 + ln] =
            acc[nt][r];
  }
}

// ---------------- flash attention with Fano mask ----------------
// 256 threads = 8 waves; block owns 128 queries (16 per wave); K/V tiles for each
// 32-key step are async-DMA'd into double-buffered LDS and shared by all 8 waves.
__global__ __launch_bounds__(256)
void fano_attn(const _Float16* __restrict__ Q, const _Float16* __restrict__ Kmat,
               const _Float16* __restrict__ VT, _Float16* __restrict__ concat,
               const int* __restrict__ causalp) {
  const int blk = blockIdx.x;            // 0..15 (128-query blocks)
  const int b = blockIdx.y, h = blockIdx.z;
  const int tid = threadIdx.x;
  const int w = tid >> 5, lane = tid & 31, g = lane >> 4, ln = lane & 15;
  const int bT = b * T_;

  const _Float16* Qh = Q    + (size_t)h * MTOK * HD;    // [8192][128]
  const _Float16* Kh = Kmat + (size_t)h * MTOK * HD;    // [8192][128]
  const _Float16* Vh = VT   + (size_t)h * HD * MTOK;    // [128][8192] (transposed)

  const int qbase = blk * 128 + w * 16;
  const bool causal = (*causalp) != 0;
  const int jmaxB = causal ? (blk * 128 + 127) : (T_ - 1);   // block-uniform bound
  const unsigned linebits =
      (1u << h) | (1u << ((h + 1) % 7)) | (1u << ((h + 3) % 7));
  const float sc2 = 0.08838834764831845f * 1.4426950408889634f; // (1/sqrt(128))*log2(e)
  const float NEG = -3.0e38f;

  // K tile: 32 keys x 128 dims, stride 136 halves (bank-rotating, 16B aligned rows)
  __shared__ __align__(16) _Float16 Ks[2 * 32 * 136];
  // V tile: 128 dims x 32 keys (from transposed V), stride 40 halves
  __shared__ __align__(16) _Float16 Vs[2 * 128 * 40];
  // per-wave P scratch for D-layout -> A-layout transpose
  __shared__ __align__(16) _Float16 Pl[8 * 16 * 40];
  _Float16* Plw = &Pl[w * 16 * 40];

  auto stage = [&](int buf, int kb) {
    // K tile: 512 x 16B chunks, 2 per thread
#pragma unroll
    for (int i = 0; i < 2; ++i) {
      int c = tid * 2 + i;                     // 0..511
      int key = c >> 4, dp = (c & 15) * 8;
      async_b128(&Ks[buf * 32 * 136 + key * 136 + dp],
                 Kh + (size_t)(bT + kb + key) * HD + dp);
    }
    // V tile: 512 x 16B chunks, 2 per thread
#pragma unroll
    for (int i = 0; i < 2; ++i) {
      int c = tid * 2 + i;                     // 0..511
      int n = c >> 2, kp = (c & 3) * 8;
      async_b128(&Vs[buf * 128 * 40 + n * 40 + kp],
                 Vh + (size_t)n * MTOK + bT + kb + kp);
    }
  };

  // preload Q operands (16 queries x 128 dims -> 4 A-operands)
  v16h qa[4];
  {
    const _Float16* qrow = Qh + (size_t)(bT + qbase + ln) * HD;
    const int off = g * 8;
#pragma unroll
    for (int dc = 0; dc < 4; ++dc) {
      v8h lo = *(const v8h*)(qrow + dc * 32 + off);
      v8h hi = *(const v8h*)(qrow + dc * 32 + 16 + off);
      qa[dc] = cat16(lo, hi);
    }
  }

  float mrow[8], lrow[8], alpha[8];
  v8f o[8];
#pragma unroll
  for (int r = 0; r < 8; ++r) { mrow[r] = NEG; lrow[r] = 0.f; }
#pragma unroll
  for (int nt = 0; nt < 8; ++nt)
#pragma unroll
    for (int r = 0; r < 8; ++r) o[nt][r] = 0.f;

  stage(0, 0);                                 // prefetch first tile

  for (int kb = 0, ib = 0; kb <= jmaxB; kb += 32, ++ib) {
    const int buf = ib & 1;
    wait_async0();
    __syncthreads();                            // tiles in LDS, prev reads done
    if (kb + 32 <= jmaxB) stage(buf ^ 1, kb + 32);   // overlap next DMA

    const _Float16* Kt = &Ks[buf * 32 * 136];
    const _Float16* Vt = &Vs[buf * 128 * 40];

    // ---- S = Q K^T for 32 keys (two 16x16 tiles) ----
    v8f s[2];
#pragma unroll
    for (int t = 0; t < 2; ++t) {
#pragma unroll
      for (int r = 0; r < 8; ++r) s[t][r] = 0.f;
      const _Float16* krow = Kt + (16 * t + ln) * 136;
#pragma unroll
      for (int dc = 0; dc < 4; ++dc) {
        v8h lo = *(const v8h*)(krow + dc * 32 + g * 8);
        v8h hi = *(const v8h*)(krow + dc * 32 + g * 8 + 16);
        s[t] = wmma16(qa[dc], cat16(lo, hi), s[t]);
      }
    }

    // ---- Fano + window + causal mask, exp2-domain scores ----
    float sm0[8], sm1[8];
#pragma unroll
    for (int t = 0; t < 2; ++t) {
      const int j  = kb + 16 * t + ln;
      const int jm = j % 7;
      const bool fan = (linebits >> jm) & 1u;
#pragma unroll
      for (int r = 0; r < 8; ++r) {
        const int i = qbase + r + 8 * g;
        bool ok = causal ? ((j <= i) && (fan || (j >= i - 16)))
                         : (fan || ((j >= i - 16) && (j <= i + 16)));
        float sv = s[t][r] * sc2;
        float* dst = t ? sm1 : sm0;
        dst[r] = ok ? sv : NEG;
      }
    }

    // ---- online softmax (rows replicated across 16-lane groups) ----
#pragma unroll
    for (int r = 0; r < 8; ++r) {
      float rm = fmaxf(sm0[r], sm1[r]);
      rm = fmaxf(rm, __shfl_xor(rm, 1, 32));
      rm = fmaxf(rm, __shfl_xor(rm, 2, 32));
      rm = fmaxf(rm, __shfl_xor(rm, 4, 32));
      rm = fmaxf(rm, __shfl_xor(rm, 8, 32));
      float mn = fmaxf(mrow[r], rm);
      float al = __builtin_exp2f(mrow[r] - mn);
      float p0 = __builtin_exp2f(sm0[r] - mn);
      float p1 = __builtin_exp2f(sm1[r] - mn);
      float rs = p0 + p1;
      rs += __shfl_xor(rs, 1, 32);
      rs += __shfl_xor(rs, 2, 32);
      rs += __shfl_xor(rs, 4, 32);
      rs += __shfl_xor(rs, 8, 32);
      lrow[r] = lrow[r] * al + rs;
      mrow[r] = mn;
      alpha[r] = al;
      Plw[(r + 8 * g) * 40 + ln]      = (_Float16)p0;  // keys kb..kb+15
      Plw[(r + 8 * g) * 40 + 16 + ln] = (_Float16)p1;  // keys kb+16..kb+31
    }
#pragma unroll
    for (int nt = 0; nt < 8; ++nt)
#pragma unroll
      for (int r = 0; r < 8; ++r) o[nt][r] *= alpha[r];

    asm volatile("s_wait_dscnt 0x0" ::: "memory");  // P stores visible to own wave

    // ---- O += P * V ----
    v16h pa;
    {
      const int off = g * 8;
      v8h lo = *(v8h*)(&Plw[ln * 40 + off]);
      v8h hi = *(v8h*)(&Plw[ln * 40 + 16 + off]);
      pa = cat16(lo, hi);
    }
#pragma unroll
    for (int nt = 0; nt < 8; ++nt) {
      const _Float16* vrow = Vt + (nt * 16 + ln) * 40 + g * 8;
      v8h lo = *(const v8h*)vrow;
      v8h hi = *(const v8h*)(vrow + 16);
      o[nt] = wmma16(pa, cat16(lo, hi), o[nt]);
    }
  }

  // ---- normalize and write per-head slice of concat buffer ----
#pragma unroll
  for (int r = 0; r < 8; ++r) {
    float inv = 1.0f / lrow[r];
    size_t rowoff = (size_t)(bT + qbase + r + 8 * g) * HID + h * HD;
#pragma unroll
    for (int nt = 0; nt < 8; ++nt)
      concat[rowoff + nt * 16 + ln] = (_Float16)(o[nt][r] * inv);
  }
}

// ---------------- host launch ----------------
extern "C" void kernel_launch(void* const* d_in, const int* in_sizes, int n_in,
                              void* d_out, int out_size, void* d_ws, size_t ws_size,
                              hipStream_t stream) {
  (void)in_sizes; (void)n_in; (void)out_size; (void)ws_size;
  const float* x  = (const float*)d_in[0];
  const float* Wq = (const float*)d_in[1];
  const float* Wk = (const float*)d_in[2];
  const float* Wv = (const float*)d_in[3];
  const float* Wo = (const float*)d_in[4];
  const int* causal = (const int*)d_in[5];

  _Float16* ws  = (_Float16*)d_ws;
  _Float16* xh  = ws + OFF_XH;
  _Float16* wqh = ws + OFF_WQ;
  _Float16* wkh = ws + OFF_WK;
  _Float16* wvh = ws + OFF_WV;
  _Float16* wot = ws + OFF_WOT;
  _Float16* qb  = ws + OFF_Q;
  _Float16* kbuf= ws + OFF_K;
  _Float16* vt  = ws + OFF_VT;
  _Float16* cc  = ws + OFF_CC;

  // 1) convert inputs/weights to f16 (WoT transposed)
  {
    int n = (int)SZ_XH;
    cvt_f32_to_f16<<<(n + 255) / 256, 256, 0, stream>>>(x, xh, n);
    n = (int)SZ_W;
    cvt_f32_to_f16<<<(n + 255) / 256, 256, 0, stream>>>(Wq, wqh, n);
    cvt_f32_to_f16<<<(n + 255) / 256, 256, 0, stream>>>(Wk, wkh, n);
    cvt_f32_to_f16<<<(n + 255) / 256, 256, 0, stream>>>(Wv, wvh, n);
    n = (int)SZ_WO;
    wo_transpose<<<(n + 255) / 256, 256, 0, stream>>>(Wo, wot);
  }

  // 2) QKV projections: per head (z), M=8192, N=128, K=896
  {
    dim3 grid(64 * 2, 1, NH), blk(256);
    size_t sB = (size_t)HID * HD;       // per-head weight stride
    size_t sO = (size_t)MTOK * HD;      // per-head output stride
    gemm16<0><<<grid, blk, 0, stream>>>(xh, wqh, sB, (void*)qb,   sO, HID, HD, HD,   2);
    gemm16<0><<<grid, blk, 0, stream>>>(xh, wkh, sB, (void*)kbuf, sO, HID, HD, HD,   2);
    gemm16<1><<<grid, blk, 0, stream>>>(xh, wvh, sB, (void*)vt,   sO, HID, HD, MTOK, 2); // V^T
  }

  // 3) flash attention with Fano mask (8 waves share async-staged K/V tiles)
  {
    dim3 grid(T_ / 128, BATCH, NH), blk(256);
    fano_attn<<<grid, blk, 0, stream>>>(qb, kbuf, vt, cc, causal);
  }

  // 4) output projection: out = concat @ Wo^T, M=8192, N=896, K=896, f32 out
  {
    dim3 grid(64 * 14, 1, 1), blk(256);
    gemm16<2><<<grid, blk, 0, stream>>>(cc, wot, 0, d_out, 0, HID, HID, HID, 14);
  }
}
